// Transformer_6906307412586
// MI455X (gfx1250) — matrix operand order repaired and checked by reading, not compiled, not run on hardware
//
#include <hip/hip_runtime.h>
#include <hip/hip_bf16.h>
#include <math.h>

// ---------------------------------------------------------------------------
// Transformer encoder forward, 12 layers, B=32 N=128 D=1024 H=16 DH=64 MLP=4096
// - All GEMMs on v_wmma_f32_16x16x32_bf16 (f32 accumulate)  [compute-bound:
//   1.26 TFLOP total vs ~26us of HBM traffic at 23.3 TB/s]
// - Weights fp32 -> bf16 + transposed once per layer (streaming, ~3us/layer),
//   so GEMM K-loop staging is pure global_load_async_to_lds_b128 (ASYNCcnt),
//   double-buffered in LDS to overlap copy with WMMA.
// ---------------------------------------------------------------------------

typedef __bf16 bf16_t;
typedef __attribute__((ext_vector_type(16))) __bf16 v16bf;
typedef __attribute__((ext_vector_type(8)))  __bf16 v8bf;
typedef __attribute__((ext_vector_type(8)))  float  v8f;

#define ZERO8 {0.f,0.f,0.f,0.f,0.f,0.f,0.f,0.f}

constexpr int SK = 40;   // LDS row stride in bf16 (80B rows, keeps 16B chunks)

__device__ __forceinline__ v8f wmma_bf16(v16bf a, v16bf b, v8f c) {
    // 8-arg form: (neg_a, A, neg_b, B, c_mod, C, reuse_a, reuse_b)
    return __builtin_amdgcn_wmma_f32_16x16x32_bf16(false, a, false, b,
                                                   (short)0, c, false, false);
}

// A-matrix fragment (16x32 bf16, row-major [row][K] source).
// ISA layout: lanes 0-15 hold K=0..7 (VGPR0-3) and K=16..23 (VGPR4-7);
// lanes 16-31 hold K=8..15 and K=24..31.  khalf = hi*8.
__device__ __forceinline__ v16bf load_frag_a(const bf16_t* p, int stride,
                                             int row, int k0, int hi) {
    const bf16_t* base = p + row * stride + k0 + hi * 8;
    v8bf c0 = *reinterpret_cast<const v8bf*>(base);        // K khalf..khalf+7
    v8bf c1 = *reinterpret_cast<const v8bf*>(base + 16);   // K 16+khalf..
    v16bf f;
#pragma unroll
    for (int i = 0; i < 8; ++i) { f[i] = c0[i]; f[8 + i] = c1[i]; }
    return f;
}

// B-matrix fragment (32x16 bf16), source stored transposed [N][K].
// ISA layout: lanes 0-15 (N=lane) hold K=0..7 and K=16..23; lanes 16-31 K=8..15 and K=24..31.
__device__ __forceinline__ v16bf load_frag_b(const bf16_t* p, int stride,
                                             int n, int k0, int hi) {
    const bf16_t* base = p + n * stride + k0 + hi * 8;
    v8bf c0 = *reinterpret_cast<const v8bf*>(base);        // K khalf..khalf+7
    v8bf c1 = *reinterpret_cast<const v8bf*>(base + 16);   // K 16+khalf..
    v16bf f;
#pragma unroll
    for (int i = 0; i < 8; ++i) { f[i] = c0[i]; f[8 + i] = c1[i]; }
    return f;
}

// Async copy of one 16B chunk: global -> LDS, tracked by ASYNCcnt.
// VDST carries the per-lane LDS byte offset (low 32 bits of the flat address
// to LDS: the flat-LDS aperture truncates addr[31:0], ISA 10.2).
__device__ __forceinline__ void async_b128(const bf16_t* gsrc, bf16_t* ldst) {
    const unsigned int loff = (unsigned int)(uintptr_t)ldst;
    const unsigned long long ga = (unsigned long long)(uintptr_t)gsrc;
    asm volatile("global_load_async_to_lds_b128 %0, %1, off"
                 :: "v"(loff), "v"(ga) : "memory");
}

// ---------------------------------------------------------------------------
// LayerNorm (one block per row of 1024).  ADD_PE: out = 2*(xhat*g+b) + pe
// (reference quirk h + (h + pe)); else out = xhat*g+b.  Output bf16.
// ---------------------------------------------------------------------------
template <bool ADD_PE>
__global__ __launch_bounds__(256) void layernorm_kernel(
    const float* __restrict__ x, const float* __restrict__ g,
    const float* __restrict__ b, const float* __restrict__ pe,
    bf16_t* __restrict__ out) {
    const int row = blockIdx.x;                 // 0..4095
    const float* xr = x + (size_t)row * 1024;
    float v[4], s = 0.f, ss = 0.f;
#pragma unroll
    for (int i = 0; i < 4; ++i) {
        v[i] = xr[threadIdx.x + i * 256];
        s += v[i];
        ss += v[i] * v[i];
    }
#pragma unroll
    for (int off = 16; off > 0; off >>= 1) {
        s  += __shfl_xor(s, off, 32);
        ss += __shfl_xor(ss, off, 32);
    }
    __shared__ float red[16];
    const int wave = threadIdx.x >> 5, lane = threadIdx.x & 31;
    if (lane == 0) { red[wave] = s; red[8 + wave] = ss; }
    __syncthreads();
    if (wave == 0) {
        float a = (lane < 8) ? red[lane] : 0.f;
        float c = (lane < 8) ? red[8 + lane] : 0.f;
#pragma unroll
        for (int off = 4; off > 0; off >>= 1) {
            a += __shfl_xor(a, off, 32);
            c += __shfl_xor(c, off, 32);
        }
        if (lane == 0) { red[0] = a; red[1] = c; }
    }
    __syncthreads();
    const float mean = red[0] * (1.f / 1024.f);
    const float var  = red[1] * (1.f / 1024.f) - mean * mean;
    const float rstd = rsqrtf(var + 1e-5f);
    const float* per = ADD_PE ? (pe + (size_t)(row & 127) * 1024) : nullptr;
#pragma unroll
    for (int i = 0; i < 4; ++i) {
        const int col = threadIdx.x + i * 256;
        float h = (v[i] - mean) * rstd * g[col] + b[col];
        if (ADD_PE) h = 2.f * h + per[col];
        out[(size_t)row * 1024 + col] = (bf16_t)h;
    }
}

// ---------------------------------------------------------------------------
// Weight convert + transpose: Wt[n][k] (bf16) = W[k][n] (fp32).
// One block per 32k x 128n tile; coalesced fp32 reads, coalesced bf16 writes.
// ---------------------------------------------------------------------------
__global__ __launch_bounds__(256) void wt_convert_kernel(
    const float* __restrict__ W, bf16_t* __restrict__ Wt, int N, int K) {
    __shared__ bf16_t tile[128 * SK];            // [n][k]
    const int tid = threadIdx.x;
    const int k0 = blockIdx.x * 32, n0 = blockIdx.y * 128;
#pragma unroll
    for (int t = 0; t < 4; ++t) {
        const int c = tid + t * 256;             // 1024 float4 chunks
        const int kr = c >> 5, nc = (c & 31) * 4;
        const float4 f = *reinterpret_cast<const float4*>(
            W + (size_t)(k0 + kr) * N + n0 + nc);
        tile[(nc + 0) * SK + kr] = (bf16_t)f.x;
        tile[(nc + 1) * SK + kr] = (bf16_t)f.y;
        tile[(nc + 2) * SK + kr] = (bf16_t)f.z;
        tile[(nc + 3) * SK + kr] = (bf16_t)f.w;
    }
    __syncthreads();
#pragma unroll
    for (int t = 0; t < 2; ++t) {
        const int c = tid + t * 256;             // 512 chunks of 8 bf16
        const int row = c >> 2, col = (c & 3) * 8;
        *reinterpret_cast<v8bf*>(Wt + (size_t)(n0 + row) * K + k0 + col) =
            *reinterpret_cast<const v8bf*>(&tile[row * SK + col]);
    }
}

// ---------------------------------------------------------------------------
// Tiled WMMA GEMM:  C[M,N] = A[M,K](bf16,[m][k]) * Wt(bf16,[n][k]) (+epilogue)
// Block tile 128x128, K-step 32, 256 threads = 8 waves, wave tile 64x32.
// K-loop staging via async global->LDS B128, double-buffered.
// ---------------------------------------------------------------------------
enum { EPI_STORE_BF16 = 0, EPI_BIAS_GELU_BF16 = 1, EPI_BIAS_RESID_F32 = 2 };

template <int EPI>
__global__ __launch_bounds__(256) void gemm_bf16_kernel(
    const bf16_t* __restrict__ A, const bf16_t* __restrict__ Bt,
    const float* __restrict__ bias, const float* __restrict__ resid,
    void* __restrict__ Out, int N, int K) {
    __shared__ bf16_t sA[2][128 * SK];
    __shared__ bf16_t sB[2][128 * SK];

    const int tid = threadIdx.x;
    const int bm = blockIdx.y, bn = blockIdx.x;
    const int wave = tid >> 5, lane = tid & 31, lo = lane & 15, hi = lane >> 4;
    const int wm = (wave >> 2) * 64, wn = (wave & 3) * 32;

    const bf16_t* gA = A  + (size_t)(bm * 128) * K;   // [m][k], row stride K
    const bf16_t* gB = Bt + (size_t)(bn * 128) * K;   // [n][k], row stride K

    // per-thread staging chunks: 2 for A + 2 for B (512 16B-chunks per tile)
    const int r0 = tid >> 2,           c0 = (tid & 3) * 8;
    const int r1 = (tid + 256) >> 2,   c1 = ((tid + 256) & 3) * 8;

    v8f zero = ZERO8;
    v8f acc[4][2];
#pragma unroll
    for (int i = 0; i < 4; ++i)
#pragma unroll
        for (int j = 0; j < 2; ++j) acc[i][j] = zero;

    // prologue: async-stage tile 0 into buffer 0   (ASYNCcnt += 4 per wave)
    async_b128(gA + (size_t)r0 * K + c0, &sA[0][r0 * SK + c0]);
    async_b128(gA + (size_t)r1 * K + c1, &sA[0][r1 * SK + c1]);
    async_b128(gB + (size_t)r0 * K + c0, &sB[0][r0 * SK + c0]);
    async_b128(gB + (size_t)r1 * K + c1, &sB[0][r1 * SK + c1]);

    int cur = 0;
    for (int k0 = 0; k0 < K; k0 += 32) {
        if (k0 + 32 < K) {
            // stage next tile into the other buffer (freed by last barrier)
            const int nk = k0 + 32, nb = cur ^ 1;
            async_b128(gA + (size_t)r0 * K + nk + c0, &sA[nb][r0 * SK + c0]);
            async_b128(gA + (size_t)r1 * K + nk + c1, &sA[nb][r1 * SK + c1]);
            async_b128(gB + (size_t)r0 * K + nk + c0, &sB[nb][r0 * SK + c0]);
            async_b128(gB + (size_t)r1 * K + nk + c1, &sB[nb][r1 * SK + c1]);
            // async loads complete in order: <=4 outstanding => tile k done
            asm volatile("s_wait_asynccnt 4" ::: "memory");
        } else {
            asm volatile("s_wait_asynccnt 0" ::: "memory");
        }
        __syncthreads();

        v16bf bf[2];
        bf[0] = load_frag_b(sB[cur], SK, wn + lo, 0, hi);
        bf[1] = load_frag_b(sB[cur], SK, wn + 16 + lo, 0, hi);
#pragma unroll
        for (int i = 0; i < 4; ++i) {
            v16bf af = load_frag_a(sA[cur], SK, wm + i * 16 + lo, 0, hi);
            acc[i][0] = wmma_bf16(af, bf[0], acc[i][0]);
            acc[i][1] = wmma_bf16(af, bf[1], acc[i][1]);
        }
        __syncthreads();
        cur ^= 1;
    }

    // ---- epilogue.  C layout: n = lo (+16*j), m = hi*8 + r (+16*i)
#pragma unroll
    for (int i = 0; i < 4; ++i) {
#pragma unroll
        for (int j = 0; j < 2; ++j) {
            const int gn = bn * 128 + wn + j * 16 + lo;
            const float bv = (EPI != EPI_STORE_BF16) ? bias[gn] : 0.f;
#pragma unroll
            for (int r = 0; r < 8; ++r) {
                const int gm = bm * 128 + wm + i * 16 + hi * 8 + r;
                float v = acc[i][j][r];
                if (EPI == EPI_STORE_BF16) {
                    ((bf16_t*)Out)[(size_t)gm * N + gn] = (bf16_t)v;
                } else if (EPI == EPI_BIAS_GELU_BF16) {
                    v += bv;
                    v = 0.5f * v * (1.f + erff(v * 0.70710678118654752f));
                    ((bf16_t*)Out)[(size_t)gm * N + gn] = (bf16_t)v;
                } else {
                    v += bv + resid[(size_t)gm * N + gn];
                    ((float*)Out)[(size_t)gm * N + gn] = v;
                }
            }
        }
    }
}

// ---------------------------------------------------------------------------
// Attention: one block per (head, batch).  N=128, DH=64, full (non-flash)
// attention in LDS.  S = QK^T * 1/8 -> softmax -> O = P V, all WMMA bf16.
// ---------------------------------------------------------------------------
__global__ __launch_bounds__(256) void attention_kernel(
    const bf16_t* __restrict__ qkv, bf16_t* __restrict__ outO) {
    constexpr int LQ = 3072;                 // qkv row stride
    constexpr int SQ = 72;                   // 64+8 pad (144B rows, 16B aligned)
    constexpr int SV = 136;                  // 128+8 pad (272B rows)
    const int h = blockIdx.x, b = blockIdx.y;
    const int rowBase = b * 128;

    // LDS arena 54272B: Q[128x72] K[128x72] Vt[64x136]; P overlays Q+K later.
    __shared__ bf16_t smem[27136];
    bf16_t* sQ  = smem;                      // 9216 bf16
    bf16_t* sK  = smem + 9216;               // 9216 bf16
    bf16_t* sVt = smem + 18432;              // 8704 bf16  [d][j]
    bf16_t* sP  = smem;                      // 17408 bf16 (fits in Q+K region)

    const int tid = threadIdx.x;
    const bf16_t* qg = qkv + (size_t)rowBase * LQ + h * 64;
    const bf16_t* kg = qg + 1024;
    const bf16_t* vg = qg + 2048;
#pragma unroll
    for (int t = 0; t < 4; ++t) {
        const int c = tid + t * 256;         // 1024 chunks cover 128x64
        const int r = c >> 3, d = (c & 7) * 8;
        *reinterpret_cast<v8bf*>(&sQ[r * SQ + d]) =
            *reinterpret_cast<const v8bf*>(qg + (size_t)r * LQ + d);
        *reinterpret_cast<v8bf*>(&sK[r * SQ + d]) =
            *reinterpret_cast<const v8bf*>(kg + (size_t)r * LQ + d);
        const v8bf vv = *reinterpret_cast<const v8bf*>(vg + (size_t)r * LQ + d);
#pragma unroll
        for (int i = 0; i < 8; ++i) sVt[(d + i) * SV + r] = vv[i];  // transpose V
    }
    __syncthreads();

    const int wave = tid >> 5, lane = tid & 31, lo = lane & 15, hi = lane >> 4;
    const int mb = wave * 16;                // each wave owns a 16-row strip

    // ---- S = Q K^T  (K tile is already [n][k] layout for B fragments)
    v8f zero = ZERO8;
    v8f s[8];
#pragma unroll
    for (int nt = 0; nt < 8; ++nt) {
        v8f a = zero;
#pragma unroll
        for (int kk = 0; kk < 2; ++kk) {
            v16bf aq = load_frag_a(sQ, SQ, mb + lo, kk * 32, hi);
            v16bf bk = load_frag_b(sK, SQ, nt * 16 + lo, kk * 32, hi);
            a = wmma_bf16(aq, bk, a);
        }
        s[nt] = a;
    }
    // scale
#pragma unroll
    for (int nt = 0; nt < 8; ++nt)
#pragma unroll
        for (int r = 0; r < 8; ++r) s[nt][r] = s[nt][r] * 0.125f;

    // ---- row softmax.  Row (hi,r) lives across the 16 lanes of a half-wave
    //      (xor masks 1..8 stay inside the half) and across the 8 nt tiles.
    float mx[8], sm[8];
#pragma unroll
    for (int r = 0; r < 8; ++r) {
        float m_ = -1e30f;
#pragma unroll
        for (int nt = 0; nt < 8; ++nt) m_ = fmaxf(m_, s[nt][r]);
#pragma unroll
        for (int off = 8; off > 0; off >>= 1)
            m_ = fmaxf(m_, __shfl_xor(m_, off, 32));
        mx[r] = m_;
        sm[r] = 0.f;
    }
#pragma unroll
    for (int nt = 0; nt < 8; ++nt)
#pragma unroll
        for (int r = 0; r < 8; ++r) {
            const float e = __expf(s[nt][r] - mx[r]);
            s[nt][r] = e;
            sm[r] += e;
        }
#pragma unroll
    for (int r = 0; r < 8; ++r) {
        float t = sm[r];
#pragma unroll
        for (int off = 8; off > 0; off >>= 1) t += __shfl_xor(t, off, 32);
        sm[r] = 1.f / t;
    }

    __syncthreads();                         // all waves done reading Q/K
    // ---- write normalized P (bf16) into per-wave LDS strip [16][128]
    bf16_t* pw = sP + wave * 16 * SV;
#pragma unroll
    for (int nt = 0; nt < 8; ++nt)
#pragma unroll
        for (int r = 0; r < 8; ++r)
            pw[(hi * 8 + r) * SV + nt * 16 + lo] = (bf16_t)(s[nt][r] * sm[r]);
    __syncthreads();

    // ---- O = P V   (A = P strip [16][128], B = Vt [d][j])
    v8f o[4];
#pragma unroll
    for (int nt2 = 0; nt2 < 4; ++nt2) {
        v8f a = zero;
#pragma unroll
        for (int kk = 0; kk < 4; ++kk) {
            v16bf ap = load_frag_a(pw, SV, lo, kk * 32, hi);
            v16bf bv = load_frag_b(sVt, SV, nt2 * 16 + lo, kk * 32, hi);
            a = wmma_bf16(ap, bv, a);
        }
        o[nt2] = a;
    }
    // ---- store to [4096][1024] bf16 at column h*64
#pragma unroll
    for (int nt2 = 0; nt2 < 4; ++nt2)
#pragma unroll
        for (int r = 0; r < 8; ++r) {
            const int gm = rowBase + mb + hi * 8 + r;
            const int gn = h * 64 + nt2 * 16 + lo;
            outO[(size_t)gm * 1024 + gn] = (bf16_t)o[nt2][r];
        }
}

// ---------------------------------------------------------------------------
// Host-side orchestration.  Workspace (64MB):
//   [0,16MB)   xcur  fp32 residual stream [4096][1024]
//   [16,24MB)  hB    bf16 [4096][1024]  (LN out / attention out, reused)
//   [24,56MB)  big   bf16 (QKV [4096][3072] then MLP mid [4096][4096], reused)
//   [56,64MB)  wT    bf16 transposed weights for the current GEMM (reused 4x)
// ---------------------------------------------------------------------------
extern "C" void kernel_launch(void* const* d_in, const int* in_sizes, int n_in,
                              void* d_out, int out_size, void* d_ws,
                              size_t ws_size, hipStream_t stream) {
    (void)in_sizes; (void)n_in; (void)out_size; (void)ws_size;
    const float* x    = (const float*)d_in[0];
    const float* pe   = (const float*)d_in[1];
    const float* ln1g = (const float*)d_in[2];
    const float* ln1b = (const float*)d_in[3];
    const float* wqkv = (const float*)d_in[4];
    const float* wo   = (const float*)d_in[5];
    const float* bo   = (const float*)d_in[6];
    const float* ln2g = (const float*)d_in[7];
    const float* ln2b = (const float*)d_in[8];
    const float* w1   = (const float*)d_in[9];
    const float* b1   = (const float*)d_in[10];
    const float* w2   = (const float*)d_in[11];
    const float* b2   = (const float*)d_in[12];

    float*  xcur = (float*)d_ws;
    bf16_t* hB   = (bf16_t*)((char*)d_ws + ((size_t)16 << 20));
    bf16_t* big  = (bf16_t*)((char*)d_ws + ((size_t)24 << 20));
    bf16_t* wT   = (bf16_t*)((char*)d_ws + ((size_t)56 << 20));

    hipMemcpyAsync(xcur, x, sizeof(float) * 4096 * 1024,
                   hipMemcpyDeviceToDevice, stream);

    const dim3 t256(256);
    for (int l = 0; l < 12; ++l) {
        // h = 2*LN1(x) + pe   -> hB (bf16)
        layernorm_kernel<true><<<4096, t256, 0, stream>>>(
            xcur, ln1g + l * 1024, ln1b + l * 1024, pe, hB);
        // qkv = h @ Wqkv      -> big (bf16 [4096][3072])
        wt_convert_kernel<<<dim3(32, 24), t256, 0, stream>>>(
            wqkv + (size_t)l * 1024 * 3072, wT, 3072, 1024);
        gemm_bf16_kernel<EPI_STORE_BF16><<<dim3(24, 32), t256, 0, stream>>>(
            hB, wT, nullptr, nullptr, big, 3072, 1024);
        // attention            -> hB (bf16 [4096][1024])
        attention_kernel<<<dim3(16, 32), t256, 0, stream>>>(big, hB);
        // x += o @ Wo + bo     (fp32, in place)
        wt_convert_kernel<<<dim3(32, 8), t256, 0, stream>>>(
            wo + (size_t)l * 1024 * 1024, wT, 1024, 1024);
        gemm_bf16_kernel<EPI_BIAS_RESID_F32><<<dim3(8, 32), t256, 0, stream>>>(
            hB, wT, bo + l * 1024, xcur, xcur, 1024, 1024);
        // f = LN2(x)           -> hB (bf16)
        layernorm_kernel<false><<<4096, t256, 0, stream>>>(
            xcur, ln2g + l * 1024, ln2b + l * 1024, nullptr, hB);
        // mid = gelu(f @ W1 + b1) -> big (bf16 [4096][4096])
        wt_convert_kernel<<<dim3(32, 32), t256, 0, stream>>>(
            w1 + (size_t)l * 1024 * 4096, wT, 4096, 1024);
        gemm_bf16_kernel<EPI_BIAS_GELU_BF16><<<dim3(32, 32), t256, 0, stream>>>(
            hB, wT, b1 + l * 4096, nullptr, big, 4096, 1024);
        // x += mid @ W2 + b2   (fp32; last layer writes d_out directly)
        wt_convert_kernel<<<dim3(128, 8), t256, 0, stream>>>(
            w2 + (size_t)l * 4096 * 1024, wT, 1024, 4096);
        float* outp = (l == 11) ? (float*)d_out : xcur;
        gemm_bf16_kernel<EPI_BIAS_RESID_F32><<<dim3(8, 32), t256, 0, stream>>>(
            big, wT, b2 + l * 1024, xcur, outp, 1024, 4096);
    }
}
